// NNUtil_76596446756983
// MI455X (gfx1250) — matrix operand order repaired and checked by reading, not compiled, not run on hardware
//
#include <hip/hip_runtime.h>

// ---------------------------------------------------------------------------
// Batched brute-force L2 kNN (B=2, M=4096, N=50000, D=32, K=32) for gfx1250.
//
// Kernel 0 (knn_preprocess): one-time split of train into f16 hi/lo arrays
//   (stored directly in WMMA-B-friendly row layout) + per-row squared norms.
//   Removes all per-tile f32->f16 conversion VALU from the hot loop (which
//   would otherwise be re-done by every one of the 512 query groups).
// Kernel 1 (knn_partial): one wave32 per (16-query group, N-slice).
//   - A tile (16 queries x 32 dims) in registers in WMMA A-layout, f32 split
//     into f16 hi+lo once per wave.
//   - Per 16-column train tile: load pre-split B (32x16) hi/lo as raw v16h,
//     3 chained v_wmma_f32_16x16x32_f16 (hi*hi + hi*lo + lo*hi).
//   - dist = t2[m] + x2[n] - 2*cross, staged through an LDS 16x20 tile so
//     each lane owns ONE query row (2 lanes/row, 8 cols each) for selection.
//   - Per-lane top-32 list kept in LDS (u64 key = sortable(dist)<<32 | col).
// Kernel 2 (knn_merge): one wave per query; 32 rounds of wave-wide u64
//   min-extraction over all partial candidates -> ascending-distance indices.
// ---------------------------------------------------------------------------

typedef __attribute__((ext_vector_type(16))) _Float16 v16h;
typedef __attribute__((ext_vector_type(8)))  float    v8f;

#define BB 2
#define MM 4096
#define NN 50000
#define DD 32
#define KK 32
#define TILES (NN / 16)          // 3125 exact
#define GROUPS ((BB * MM) / 16)  // 512
#define QTOT (BB * MM)           // 8192
#define ROWS (BB * NN)           // 100000 train rows

static __device__ __forceinline__ unsigned sortable_f32(float f) {
  unsigned u = __float_as_uint(f);
  return u ^ ((unsigned)((int)u >> 31) | 0x80000000u);
}

// ---- Kernel 0: split train f32 -> f16 hi/lo + row squared norms ----
__global__ void __launch_bounds__(256) knn_preprocess(
    const float* __restrict__ train, _Float16* __restrict__ hiOut,
    _Float16* __restrict__ loOut, float* __restrict__ x2Out) {
  const int row = blockIdx.x * 256 + threadIdx.x;
  if (row >= ROWS) return;
  const float4* src = (const float4*)(train + (size_t)row * DD);
  float fa[32];
#pragma unroll
  for (int v = 0; v < 8; ++v) {
    float4 q = src[v];
    fa[v * 4 + 0] = q.x; fa[v * 4 + 1] = q.y;
    fa[v * 4 + 2] = q.z; fa[v * 4 + 3] = q.w;
  }
  union { float4 f4[4]; _Float16 h[32]; } uh, ul;
  float s = 0.f;
#pragma unroll
  for (int k = 0; k < 32; ++k) {
    float x = fa[k];
    s += x * x;
    _Float16 hi = (_Float16)x;
    uh.h[k] = hi;
    ul.h[k] = (_Float16)(x - (float)hi);
  }
  float4* dh = (float4*)(hiOut + (size_t)row * DD);
  float4* dl = (float4*)(loOut + (size_t)row * DD);
#pragma unroll
  for (int v = 0; v < 4; ++v) { dh[v] = uh.f4[v]; dl[v] = ul.f4[v]; }
  x2Out[row] = s;
}

// ---- Kernel 1: fused WMMA distance + per-lane top-32 selection ----
__global__ void __launch_bounds__(128) knn_partial(
    const float* __restrict__ test, const _Float16* __restrict__ trainHi,
    const _Float16* __restrict__ trainLo, const float* __restrict__ x2arr,
    unsigned long long* __restrict__ ws, int S) {
  __shared__ float tileD[4][16 * 20];             // 16x16 dist tile, stride 20
  __shared__ float t2s[4][16];                    // per-row |test|^2
  __shared__ unsigned long long cand[4][32 * 32]; // per-lane top-32 lists

  const int lane = threadIdx.x & 31;
  const int wib  = threadIdx.x >> 5;
  const int wglobal = blockIdx.x * 4 + wib;
  const int group = wglobal / S;   // [0, 512)
  const int slice = wglobal % S;
  if (group >= GROUPS) return;     // grid is exact; uniform branch

  const int b  = group / (MM / 16);
  const int m0 = (group % (MM / 16)) * 16;

  const float*    testB    = test + ((size_t)b * MM + m0) * DD;
  const _Float16* trainHiB = trainHi + (size_t)b * NN * DD;
  const _Float16* trainLoB = trainLo + (size_t)b * NN * DD;
  const float*    x2B      = x2arr + (size_t)b * NN;

  // ---- t2 (query squared norms) into LDS ----
  if (lane < 16) {
    const float4* row = (const float4*)(testB + (size_t)lane * DD);
    float s = 0.f;
#pragma unroll
    for (int v = 0; v < 8; ++v) {
      float4 q = row[v];
      s += q.x * q.x + q.y * q.y + q.z * q.z + q.w * q.w;
    }
    t2s[wib][lane] = s;
  }
  asm volatile("s_wait_dscnt 0" ::: "memory");

  const int hiHalf = (lane >= 16) ? 1 : 0;
  float t2v[8];
#pragma unroll
  for (int r = 0; r < 8; ++r) t2v[r] = t2s[wib][r + hiHalf * 8];

  // ---- A (16x32 test tile) in WMMA A-layout, f16 hi/lo split (once) ----
  // half h of lane l -> K = (l<16 ? 0 : 16) + h, M = l%16
  v16h a_hi, a_lo;
  {
    const float* arow = testB + (size_t)(lane & 15) * DD;
    const int koff = hiHalf * 16;
    float fa[16];
    {
      const float4* p0 = (const float4*)(arow + koff);
      const float4* p1 = (const float4*)(arow + koff + 8);
      float4 r0 = p0[0], r1 = p0[1], r2 = p1[0], r3 = p1[1];
      fa[0]=r0.x; fa[1]=r0.y; fa[2]=r0.z; fa[3]=r0.w;
      fa[4]=r1.x; fa[5]=r1.y; fa[6]=r1.z; fa[7]=r1.w;
      fa[8]=r2.x; fa[9]=r2.y; fa[10]=r2.z; fa[11]=r2.w;
      fa[12]=r3.x; fa[13]=r3.y; fa[14]=r3.z; fa[15]=r3.w;
    }
#pragma unroll
    for (int h = 0; h < 16; ++h) {
      float x = fa[h];
      _Float16 hi = (_Float16)x;
      a_hi[h] = hi;
      a_lo[h] = (_Float16)(x - (float)hi);
    }
  }

  // ---- init per-lane candidate list (LDS) ----
  unsigned long long* mycand = &cand[wib][lane * 32];
#pragma unroll
  for (int e = 0; e < 32; ++e) mycand[e] = ~0ull;
  unsigned long long worstKey = ~0ull;
  int worstPos = 0;

  const int rRow = lane >> 1;         // selection: my query row
  const int rCol = (lane & 1) * 8;    // my 8 columns within tile

  for (int t = slice; t < TILES; t += S) {
    const int n0 = t * 16;
    const size_t bOff = (size_t)(n0 + (lane & 15)) * DD + hiHalf * 16;

    if (t + S < TILES) {
      const size_t pOff = (size_t)(n0 + 16 * S + (lane & 15)) * DD;
      __builtin_prefetch(trainHiB + pOff, 0, 1);
      __builtin_prefetch(trainLoB + pOff, 0, 1);
    }

    // ---- B (32x16 train tile) pre-split, raw v16h loads (32B each) ----
    // half h -> K = (l<16 ? 0 : 16) + h, N = l%16
    const v16h b_hi = *(const v16h*)(trainHiB + bOff);
    const v16h b_lo = *(const v16h*)(trainLoB + bOff);
    const float x2 = x2B[n0 + (lane & 15)];

    // ---- cross = A.B via 3 split-precision WMMAs ----
    v8f c = {};
    c = __builtin_amdgcn_wmma_f32_16x16x32_f16(false, a_hi, false, b_hi,
                                               (short)0, c, false, false);
    c = __builtin_amdgcn_wmma_f32_16x16x32_f16(false, a_hi, false, b_lo,
                                               (short)0, c, false, false);
    c = __builtin_amdgcn_wmma_f32_16x16x32_f16(false, a_lo, false, b_hi,
                                               (short)0, c, false, false);

    // ---- distances -> LDS tile (row = r + hiHalf*8, col = l%16) ----
    float* td = tileD[wib];
#pragma unroll
    for (int r = 0; r < 8; ++r) {
      float d = t2v[r] + x2 - 2.0f * c[r];
      td[(r + hiHalf * 8) * 20 + (lane & 15)] = d;
    }
    asm volatile("s_wait_dscnt 0" ::: "memory");

    // ---- transpose read: lane owns one row, 8 contiguous cols ----
    float dr[8];
    {
      const float4* pr = (const float4*)&td[rRow * 20 + rCol];
      float4 d0 = pr[0], d1 = pr[1];
      dr[0]=d0.x; dr[1]=d0.y; dr[2]=d0.z; dr[3]=d0.w;
      dr[4]=d1.x; dr[5]=d1.y; dr[6]=d1.z; dr[7]=d1.w;
    }
    asm volatile("s_wait_dscnt 0" ::: "memory");  // WAR vs next tile's stores

    // ---- top-32 maintenance (threshold + replace-worst) ----
#pragma unroll
    for (int j = 0; j < 8; ++j) {
      unsigned long long key =
          ((unsigned long long)sortable_f32(dr[j]) << 32) |
          (unsigned)(n0 + rCol + j);
      if (key < worstKey) {
        mycand[worstPos] = key;
        unsigned long long wk = 0ull;
        int wp = 0;
#pragma unroll
        for (int e = 0; e < 32; ++e) {
          unsigned long long v = mycand[e];
          if (v > wk) { wk = v; wp = e; }
        }
        worstKey = wk;
        worstPos = wp;
      }
    }
  }

  // ---- dump partial list: ws[q][slice][sublane][32] ----
  const int q = group * 16 + rRow;
  unsigned long long* dst =
      ws + ((((size_t)q * S + slice) * 2 + (lane & 1)) * 32);
#pragma unroll
  for (int e = 0; e < 32; ++e) dst[e] = mycand[e];
}

// ---- Kernel 2: wave-per-query merge of partial top-32 lists ----
__global__ void __launch_bounds__(256) knn_merge(
    const unsigned long long* __restrict__ ws, long long* __restrict__ out,
    int S) {
  const int lane = threadIdx.x & 31;
  const int wib  = threadIdx.x >> 5;
  const int q = blockIdx.x * 8 + wib;
  if (q >= QTOT) return;

  const int E = S * 64;  // candidates per query (<= 512)
  const unsigned long long* src = ws + (size_t)q * (size_t)E;

  unsigned long long local[16];
#pragma unroll
  for (int i = 0; i < 16; ++i) {
    int e = i * 32 + lane;
    local[i] = (e < E) ? src[e] : ~0ull;
  }

  unsigned used = 0;
  for (int iter = 0; iter < KK; ++iter) {
    unsigned long long mn = ~0ull;
#pragma unroll
    for (int i = 0; i < 16; ++i) {
      unsigned long long v = ((used >> i) & 1u) ? ~0ull : local[i];
      mn = (v < mn) ? v : mn;
    }
    unsigned long long wmn = mn;
#pragma unroll
    for (int off = 16; off >= 1; off >>= 1) {
      unsigned long long o = __shfl_xor(wmn, off, 32);
      wmn = (o < wmn) ? o : wmn;
    }
    if (mn == wmn) {  // owner lane retires its entry (keys are unique)
      bool done = false;
#pragma unroll
      for (int i = 0; i < 16; ++i) {
        if (!done && !((used >> i) & 1u) && local[i] == wmn) {
          used |= (1u << i);
          done = true;
        }
      }
    }
    if (lane == 0)
      out[(size_t)q * KK + iter] = (long long)(unsigned)(wmn & 0xFFFFFFFFull);
  }
}

extern "C" void kernel_launch(void* const* d_in, const int* in_sizes, int n_in,
                              void* d_out, int out_size, void* d_ws,
                              size_t ws_size, hipStream_t stream) {
  (void)in_sizes; (void)n_in; (void)out_size;
  const float* test  = (const float*)d_in[0];   // (B, M, D) f32
  const float* train = (const float*)d_in[1];   // (B, N, D) f32
  long long* out = (long long*)d_out;           // (B, M, K) int64

  // ws layout: trainHi (6.4MB) | trainLo (6.4MB) | x2 (0.4MB) | partials
  char* base = (char*)d_ws;
  const size_t hiBytes = (size_t)ROWS * DD * sizeof(_Float16);  // 6,400,000
  const size_t x2Bytes = ((size_t)ROWS * sizeof(float) + 255) & ~(size_t)255;
  _Float16* trainHi = (_Float16*)base;
  _Float16* trainLo = (_Float16*)(base + hiBytes);
  float*    x2arr   = (float*)(base + 2 * hiBytes);
  unsigned long long* part =
      (unsigned long long*)(base + 2 * hiBytes + x2Bytes);
  const size_t fixedBytes = 2 * hiBytes + x2Bytes;

  // Each N-slice of partial results needs Q * 64 entries * 8B = 4 MiB.
  const size_t perSlice = (size_t)QTOT * 64 * sizeof(unsigned long long);
  int S = 1;
  if (ws_size > fixedBytes + perSlice)
    S = (int)((ws_size - fixedBytes) / perSlice);
  if (S < 1) S = 1;
  if (S > 8) S = 8;

  dim3 blkP(256), grdP((ROWS + 255) / 256);
  knn_preprocess<<<grdP, blkP, 0, stream>>>(train, trainHi, trainLo, x2arr);

  dim3 blk1(128), grd1(128 * S);  // 512*S waves, 4 waves/block
  knn_partial<<<grd1, blk1, 0, stream>>>(test, trainHi, trainLo, x2arr, part,
                                         S);

  dim3 blk2(256), grd2(QTOT / 8);
  knn_merge<<<grd2, blk2, 0, stream>>>(part, out, S);
}
